// MultiHeadDotProductAttentionWithWeight_44547400794560
// MI455X (gfx1250) — matrix operand order repaired and checked by reading, not compiled, not run on hardware
//
#include <hip/hip_runtime.h>

typedef __bf16 bf16;
typedef __attribute__((ext_vector_type(4)))  float  f32x4;
typedef __attribute__((ext_vector_type(8)))  float  f32x8;
typedef __attribute__((ext_vector_type(4)))  __bf16 bf16x4;
typedef __attribute__((ext_vector_type(8)))  __bf16 bf16x8;
typedef __attribute__((ext_vector_type(16))) __bf16 bf16x16;
typedef __attribute__((ext_vector_type(8)))  float  v8f;
typedef int i32x4 __attribute__((vector_size(4 * sizeof(int))));

#define BIG_NEG (-3.402823466e38f)

// ---------------- gfx1250 async global->LDS copy (ASYNCcnt-tracked) ----------

#if defined(__has_builtin)
#if __has_builtin(__builtin_amdgcn_global_load_async_to_lds_b128)
#define HAVE_ASYNC_LDS 1
#endif
#endif
#ifndef HAVE_ASYNC_LDS
#define HAVE_ASYNC_LDS 0
#endif

static __device__ inline void stage_b128(const bf16* gsrc, bf16* ldst) {
#if HAVE_ASYNC_LDS
  __builtin_amdgcn_global_load_async_to_lds_b128(
      (__attribute__((address_space(1))) i32x4*)gsrc,
      (__attribute__((address_space(3))) i32x4*)ldst, 0, 0);
#else
  *(bf16x8*)ldst = *(const bf16x8*)gsrc;
#endif
}

static __device__ inline void stage_wait() {
#if HAVE_ASYNC_LDS
#if __has_builtin(__builtin_amdgcn_s_wait_asynccnt)
  __builtin_amdgcn_s_wait_asynccnt(0);
#else
  asm volatile("s_wait_asynccnt 0x0" ::: "memory");
#endif
#endif
}

// ---------------- WMMA fragment helpers (CDNA5 wave32 layouts) ----------------

// A-matrix 16x32 bf16: lanes 0-15 row m=lane hold K 0..15 contiguous,
// lanes 16-31 row m=lane-16 hold K 16..31.
static __device__ inline bf16x16 fragA_bf16(const bf16* p0, int ld) {
  const int lane = threadIdx.x & 31;
  const int half = lane >> 4;
  const int r    = lane & 15;
  const bf16* p = p0 + r * ld + half * 16;
  bf16x8 lo = *(const bf16x8*)(p);        // K = 16*half .. 16*half+7
  bf16x8 hi = *(const bf16x8*)(p + 8);    // K = 16*half+8 .. 16*half+15
  return __builtin_shufflevector(lo, hi, 0,1,2,3,4,5,6,7,8,9,10,11,12,13,14,15);
}

// Same A fragment but sourced from f32 (softmax weights), converted to bf16.
static __device__ inline bf16x16 fragA_f32(const float* p0, int ld) {
  const int lane = threadIdx.x & 31;
  const int half = lane >> 4;
  const int r    = lane & 15;
  const float* p = p0 + r * ld + half * 16;
  f32x8 lo = *(const f32x8*)(p);
  f32x8 hi = *(const f32x8*)(p + 8);
  bf16x8 l = __builtin_convertvector(lo, bf16x8);
  bf16x8 h = __builtin_convertvector(hi, bf16x8);
  return __builtin_shufflevector(l, h, 0,1,2,3,4,5,6,7,8,9,10,11,12,13,14,15);
}

// B-matrix 32x16 bf16: lane group 0 (lanes 0-15) col n=lane holds K 0..15
// contiguous, lane group 1 holds K 16..31. p0 points at [n=0,k=0] of a tile
// whose K dim is contiguous with stride ld between columns.
static __device__ inline bf16x16 fragB_bf16(const bf16* p0, int ld) {
  const int lane = threadIdx.x & 31;
  const int half = lane >> 4;
  const int n    = lane & 15;
  return *(const bf16x16*)(p0 + n * ld + half * 16);
}

static __device__ inline v8f wmma_bf16(bf16x16 a, bf16x16 b, v8f c) {
  return __builtin_amdgcn_wmma_f32_16x16x32_bf16(false, a, false, b, (short)0, c,
                                                 false, false);
}

static __device__ inline v8f v8f_zero() {
  v8f z = {0.0f, 0.0f, 0.0f, 0.0f, 0.0f, 0.0f, 0.0f, 0.0f};
  return z;
}

// ---------------- f32 -> bf16 cast (vectorized) ----------------

__global__ void __launch_bounds__(256) cvt_bf16_kernel(const float* __restrict__ in,
                                                       bf16* __restrict__ out, int n4) {
  int i = blockIdx.x * 256 + threadIdx.x;
  if (i < n4) {
    f32x4 v = ((const f32x4*)in)[i];
    ((bf16x4*)out)[i] = __builtin_convertvector(v, bf16x4);
  }
}

// ---------------- Generic bf16 GEMM: C = alpha*(A[MxK] * W[KxN] + bias) ------
// Block tile 128x64, 8 waves, each wave 32x32 (2x2 WMMA frags). K step 32.

__global__ void __launch_bounds__(256) gemm_bf16_kernel(
    const bf16* __restrict__ A, const bf16* __restrict__ W,
    const float* __restrict__ bias, float alpha,
    float* __restrict__ Cf, bf16* __restrict__ Cb,
    int M, int N, int K) {
  __shared__ bf16 As[128 * 32];   // row-major [m][k]
  __shared__ bf16 Wt[64 * 32];    // transposed [n][k]
  const int tid  = threadIdx.x;
  const int lane = tid & 31;
  const int wave = tid >> 5;
  const int wr = wave & 3, wc = wave >> 2;
  const int m0 = blockIdx.x * 128, n0 = blockIdx.y * 64;

  v8f acc00 = v8f_zero(), acc01 = v8f_zero(), acc10 = v8f_zero(), acc11 = v8f_zero();

  for (int k0 = 0; k0 < K; k0 += 32) {
    // stage A tile 128x32: async DMA global->LDS (two b128 per thread)
    {
      int idx = tid;
      for (int t = 0; t < 2; ++t, idx += 256) {
        int r = idx >> 2, c = (idx & 3) * 8;
        stage_b128(&A[(size_t)(m0 + r) * K + k0 + c], &As[r * 32 + c]);
      }
    }
    // stage W tile 32x64, transposed into Wt[n][k] (needs VALU, stays sync)
    {
      int kk = tid >> 3, nn = (tid & 7) * 8;
      bf16x8 w = *(const bf16x8*)&W[(size_t)(k0 + kk) * N + n0 + nn];
      for (int e = 0; e < 8; ++e) Wt[(nn + e) * 32 + kk] = w[e];
    }
    if (k0 + 32 < K)  // pull next A slab toward near caches (global_prefetch)
      __builtin_prefetch(&A[(size_t)(m0 + (tid >> 2)) * K + k0 + 32], 0, 3);
    stage_wait();
    __syncthreads();

    bf16x16 a0 = fragA_bf16(&As[(wr * 32 + 0) * 32], 32);
    bf16x16 a1 = fragA_bf16(&As[(wr * 32 + 16) * 32], 32);
    bf16x16 b0 = fragB_bf16(&Wt[(wc * 32 + 0) * 32], 32);
    bf16x16 b1 = fragB_bf16(&Wt[(wc * 32 + 16) * 32], 32);
    acc00 = wmma_bf16(a0, b0, acc00);
    acc01 = wmma_bf16(a0, b1, acc01);
    acc10 = wmma_bf16(a1, b0, acc10);
    acc11 = wmma_bf16(a1, b1, acc11);
    __syncthreads();
  }

  const int half = lane >> 4, ln = lane & 15;
  v8f accs[2][2] = {{acc00, acc01}, {acc10, acc11}};
  for (int i = 0; i < 2; ++i)
    for (int j = 0; j < 2; ++j) {
      int n = n0 + wc * 32 + j * 16 + ln;
      float bn = bias ? bias[n] : 0.0f;
      for (int r = 0; r < 8; ++r) {
        int m = m0 + wr * 32 + i * 16 + r + 8 * half;
        float v = alpha * (accs[i][j][r] + bn);
        if (Cb) Cb[(size_t)m * N + n] = (bf16)v;
        else    Cf[(size_t)m * N + n] = v;
      }
    }
}

// ---------------- Scores: attn[bh,i,j] = Q[b,i,h,:].K[b,j,h,:] + causal mask --
// Block tile 128(i) x 64(j) per (b,h). K^T fragments are contiguous along d.

__global__ void __launch_bounds__(256) attn_scores_kernel(
    const bf16* __restrict__ Q, const bf16* __restrict__ Kmat,
    float* __restrict__ attn, int S, int HD, int H, int Dh) {
  const int tid = threadIdx.x, lane = tid & 31, wave = tid >> 5;
  const int wr = wave & 3, wc = wave >> 2;
  const int i0 = blockIdx.x * 128, j0 = blockIdx.y * 64;
  const int bh = blockIdx.z, b = bh / H, h = bh % H;
  float* arow = attn + (size_t)bh * S * S;

  if (j0 >= i0 + 128) {  // fully masked block: fill and exit, no GEMM
    for (int t = tid; t < 128 * 64; t += 256) {
      int r = t >> 6, c = t & 63;
      arow[(size_t)(i0 + r) * S + (j0 + c)] = BIG_NEG;
    }
    return;
  }

  const bf16* Qb = Q + ((size_t)b * S) * HD + h * Dh;
  const bf16* Kb = Kmat + ((size_t)b * S) * HD + h * Dh;

  v8f acc00 = v8f_zero(), acc01 = v8f_zero(), acc10 = v8f_zero(), acc11 = v8f_zero();
  for (int d0 = 0; d0 < Dh; d0 += 32) {
    bf16x16 a0 = fragA_bf16(Qb + (size_t)(i0 + wr * 32 + 0) * HD + d0, HD);
    bf16x16 a1 = fragA_bf16(Qb + (size_t)(i0 + wr * 32 + 16) * HD + d0, HD);
    bf16x16 b0 = fragB_bf16(Kb + (size_t)(j0 + wc * 32 + 0) * HD + d0, HD);
    bf16x16 b1 = fragB_bf16(Kb + (size_t)(j0 + wc * 32 + 16) * HD + d0, HD);
    acc00 = wmma_bf16(a0, b0, acc00);
    acc01 = wmma_bf16(a0, b1, acc01);
    acc10 = wmma_bf16(a1, b0, acc10);
    acc11 = wmma_bf16(a1, b1, acc11);
  }

  const int half = lane >> 4, ln = lane & 15;
  v8f accs[2][2] = {{acc00, acc01}, {acc10, acc11}};
  for (int i = 0; i < 2; ++i)
    for (int j = 0; j < 2; ++j) {
      int n = j0 + wc * 32 + j * 16 + ln;
      for (int r = 0; r < 8; ++r) {
        int m = i0 + wr * 32 + i * 16 + r + 8 * half;
        float v = accs[i][j][r];
        if (n > m) v = BIG_NEG;               // causal mask
        arow[(size_t)m * S + n] = v;
      }
    }
}

// ---------------- Row softmax in place (one 256-thread block per row) --------

__global__ void __launch_bounds__(256) softmax_rows_kernel(float* __restrict__ attn, int S) {
  __shared__ float red[256];
  const int tid = threadIdx.x;
  float* p = attn + (size_t)blockIdx.x * S;   // S == 2048 == 8*256
  float vals[8];
  float lm = BIG_NEG;
  for (int t = 0; t < 8; ++t) { vals[t] = p[tid + t * 256]; lm = fmaxf(lm, vals[t]); }
  red[tid] = lm; __syncthreads();
  for (int s = 128; s > 0; s >>= 1) { if (tid < s) red[tid] = fmaxf(red[tid], red[tid + s]); __syncthreads(); }
  float rowmax = red[0]; __syncthreads();
  float ls = 0.0f;
  for (int t = 0; t < 8; ++t) { vals[t] = __expf(vals[t] - rowmax); ls += vals[t]; }
  red[tid] = ls; __syncthreads();
  for (int s = 128; s > 0; s >>= 1) { if (tid < s) red[tid] += red[tid + s]; __syncthreads(); }
  float inv = 1.0f / red[0];
  for (int t = 0; t < 8; ++t) p[tid + t * 256] = vals[t] * inv;
}

// ---------------- PV: X[b,i,h,d] = sum_j P[bh,i,j] * V[b,j,h,d] --------------
// Block: 128 rows x 64 (=Dh) cols per (b,h); causal j-loop stops at i0+128.

__global__ void __launch_bounds__(256) attn_pv_kernel(
    const float* __restrict__ attn, const bf16* __restrict__ V,
    bf16* __restrict__ X, int S, int HD, int H, int Dh) {
  __shared__ bf16 Vt[64 * 32];   // [d][j]
  const int tid = threadIdx.x, lane = tid & 31, wave = tid >> 5;
  const int wr = wave & 3, wc = wave >> 2;
  const int i0 = blockIdx.x * 128;
  const int bh = blockIdx.y, b = bh / H, h = bh % H;
  const float* P = attn + (size_t)bh * S * S;
  const bf16* Vb = V + ((size_t)b * S) * HD + h * Dh;

  v8f acc00 = v8f_zero(), acc01 = v8f_zero(), acc10 = v8f_zero(), acc11 = v8f_zero();

  for (int j0 = 0; j0 < i0 + 128; j0 += 32) {
    {  // stage V chunk 32x64 transposed: Vt[d][j] = V[j0+j][d]
      int jj = tid >> 3, d8 = (tid & 7) * 8;
      bf16x8 w = *(const bf16x8*)&Vb[(size_t)(j0 + jj) * HD + d8];
      for (int e = 0; e < 8; ++e) Vt[(d8 + e) * 32 + jj] = w[e];
    }
    __syncthreads();
    bf16x16 a0 = fragA_f32(P + (size_t)(i0 + wr * 32 + 0) * S + j0, S);
    bf16x16 a1 = fragA_f32(P + (size_t)(i0 + wr * 32 + 16) * S + j0, S);
    bf16x16 b0 = fragB_bf16(&Vt[(wc * 32 + 0) * 32], 32);
    bf16x16 b1 = fragB_bf16(&Vt[(wc * 32 + 16) * 32], 32);
    acc00 = wmma_bf16(a0, b0, acc00);
    acc01 = wmma_bf16(a0, b1, acc01);
    acc10 = wmma_bf16(a1, b0, acc10);
    acc11 = wmma_bf16(a1, b1, acc11);
    __syncthreads();
  }

  const int half = lane >> 4, ln = lane & 15;
  v8f accs[2][2] = {{acc00, acc01}, {acc10, acc11}};
  for (int i = 0; i < 2; ++i)
    for (int j = 0; j < 2; ++j) {
      int d = wc * 32 + j * 16 + ln;
      for (int r = 0; r < 8; ++r) {
        int m = i0 + wr * 32 + i * 16 + r + 8 * half;   // position in S
        X[((size_t)b * S + m) * HD + h * Dh + d] = (bf16)accs[i][j][r];
      }
    }
}

// ---------------- Host orchestration ----------------

extern "C" void kernel_launch(void* const* d_in, const int* in_sizes, int n_in,
                              void* d_out, int out_size, void* d_ws, size_t ws_size,
                              hipStream_t stream) {
  (void)in_sizes; (void)n_in; (void)out_size; (void)ws_size;
  constexpr int Bn = 2, S = 2048, F = 1024, H = 16, Dh = 64;
  constexpr int M = Bn * S;      // 4096 rows
  constexpr int HD = H * Dh;     // 1024

  const float* inputs_q  = (const float*)d_in[0];
  const float* inputs_kv = (const float*)d_in[1];
  // d_in[2] = mask (causal; implemented analytically)
  const float* Wq = (const float*)d_in[3];
  const float* bq = (const float*)d_in[4];
  const float* Wk = (const float*)d_in[5];
  const float* bk = (const float*)d_in[6];
  const float* Wv = (const float*)d_in[7];
  const float* bv = (const float*)d_in[8];
  const float* Wo = (const float*)d_in[9];
  const float* bo = (const float*)d_in[10];

  char* ws = (char*)d_ws;
  bf16* Xq  = (bf16*)ws; ws += (size_t)M * F * 2;
  bf16* Xkv = (bf16*)ws; ws += (size_t)M * F * 2;
  bf16* Wqb = (bf16*)ws; ws += (size_t)F * HD * 2;
  bf16* Wkb = (bf16*)ws; ws += (size_t)F * HD * 2;
  bf16* Wvb = (bf16*)ws; ws += (size_t)F * HD * 2;
  bf16* Wob = (bf16*)ws; ws += (size_t)HD * F * 2;
  bf16* Qb  = (bf16*)ws; ws += (size_t)M * HD * 2;
  bf16* Kb  = (bf16*)ws; ws += (size_t)M * HD * 2;
  bf16* Vb  = (bf16*)ws; ws += (size_t)M * HD * 2;
  bf16* Xb  = (bf16*)ws; ws += (size_t)M * HD * 2;

  float* out  = (float*)d_out;
  float* attn = out + (size_t)M * F;   // attn_weights region of d_out

  auto cvt = [&](const float* src, bf16* dst, int n) {
    int n4 = n / 4;
    cvt_bf16_kernel<<<(n4 + 255) / 256, 256, 0, stream>>>(src, dst, n4);
  };
  cvt(inputs_q,  Xq,  M * F);
  cvt(inputs_kv, Xkv, M * F);
  cvt(Wq, Wqb, F * HD);
  cvt(Wk, Wkb, F * HD);
  cvt(Wv, Wvb, F * HD);
  cvt(Wo, Wob, HD * F);

  dim3 gproj(M / 128, HD / 64);   // 32 x 16
  // Q scaled by 1/sqrt(Dh) = 0.125 (applied to acc+bias, matching reference)
  gemm_bf16_kernel<<<gproj, 256, 0, stream>>>(Xq,  Wqb, bq, 0.125f, nullptr, Qb, M, HD, F);
  gemm_bf16_kernel<<<gproj, 256, 0, stream>>>(Xkv, Wkb, bk, 1.0f,   nullptr, Kb, M, HD, F);
  gemm_bf16_kernel<<<gproj, 256, 0, stream>>>(Xkv, Wvb, bv, 1.0f,   nullptr, Vb, M, HD, F);

  dim3 gs(S / 128, S / 64, Bn * H);  // 16 x 32 x 32
  attn_scores_kernel<<<gs, 256, 0, stream>>>(Qb, Kb, attn, S, HD, H, Dh);

  softmax_rows_kernel<<<Bn * H * S, 256, 0, stream>>>(attn, S);

  dim3 gp(S / 128, Bn * H);          // 16 x 32
  attn_pv_kernel<<<gp, 256, 0, stream>>>(attn, Vb, Xb, S, HD, H, Dh);

  gemm_bf16_kernel<<<gproj, 256, 0, stream>>>(Xb, Wob, bo, 1.0f, out, nullptr, M, F, HD);
}
